// Encoder_8237747274312
// MI455X (gfx1250) — matrix operand order repaired and checked by reading, not compiled, not run on hardware
//
#include <hip/hip_runtime.h>

// ---------------- problem constants ----------------
#define BATCH  64
#define SEQ    512
#define HID    300
#define UNITS  512
#define G3     1536            // 3*UNITS
#define NT     96              // G3/16 column tiles
#define MT     4               // BATCH/16 row tiles
#define KT_H   16              // UNITS/32 k-tiles
#define KT_X0  10              // ceil(300/32)=10, K padded to 320
// ---------------- launch shape ----------------
#define NWG       32
#define TPB       384          // 12 waves/WG -> 384 waves == NT*MT tile jobs (1 tile/wave)
#define NTHREADS  (NWG*TPB)
#define NWAVES    (NTHREADS/32)

typedef __attribute__((ext_vector_type(16))) _Float16 v16h;
typedef __attribute__((ext_vector_type(4)))  _Float16 v4h;
typedef __attribute__((ext_vector_type(8)))  float    v8f;

__device__ __forceinline__ float sigmoidf_(float x) { return 1.0f / (1.0f + __expf(-x)); }

__device__ __forceinline__ float gru1(float gz, float gr, float xh, float rh, float h) {
  float z  = sigmoidf_(gz);
  float r  = sigmoidf_(gr);
  float hh = tanhf(xh + r * rh);
  return z * h + (1.0f - z) * hh;
}

// ---------------- device-wide sense barrier ----------------
__device__ __forceinline__ void gbar(unsigned* bar) {
  __syncthreads();
  if (threadIdx.x == 0) {
    __threadfence();
    unsigned g = __hip_atomic_load(&bar[1], __ATOMIC_RELAXED, __HIP_MEMORY_SCOPE_AGENT);
    unsigned a = __hip_atomic_fetch_add(&bar[0], 1u, __ATOMIC_ACQ_REL, __HIP_MEMORY_SCOPE_AGENT);
    if (a == NWG - 1u) {
      __hip_atomic_store(&bar[0], 0u, __ATOMIC_RELAXED, __HIP_MEMORY_SCOPE_AGENT);
      __hip_atomic_fetch_add(&bar[1], 1u, __ATOMIC_RELEASE, __HIP_MEMORY_SCOPE_AGENT);
    } else {
      while (__hip_atomic_load(&bar[1], __ATOMIC_ACQUIRE, __HIP_MEMORY_SCOPE_AGENT) == g)
        __builtin_amdgcn_s_sleep(2);
    }
    __threadfence();
  }
  __syncthreads();
}

// ---------------- weight pack: f32 [K,1536] row-major -> f16 B-operand layout ----------------
// B 32x16 f16 (KxN): col n=lane%16; slots 0..7 hold K=8*half+0..7, slots 8..15 hold K=16+8*half+0..7.
__global__ void pack_weights(const float* __restrict__ src, _Float16* __restrict__ dst,
                             int Ksrc, int KT) {
  int total = NT * KT * 512;
  for (int d = blockIdx.x * blockDim.x + threadIdx.x; d < total; d += gridDim.x * blockDim.x) {
    int nt   = d / (KT * 512);
    int r    = d - nt * (KT * 512);
    int kt   = r >> 9;
    int q    = r & 511;
    int lane = q >> 4;
    int slot = q & 15;
    int n = nt * 16 + (lane & 15);
    int k = kt * 32 + 8 * (lane >> 4) + (slot & 7) + 16 * (slot >> 3);
    float v = (k < Ksrc) ? src[(size_t)k * G3 + n] : 0.0f;
    dst[d] = (_Float16)v;
  }
}

// ---------------- init: zero h state (f32 + packed f16), reset barrier ----------------
__global__ void init_state(float* h0f, float* h1f, _Float16* h0A, _Float16* h1A, unsigned* bar) {
  int tid = blockIdx.x * blockDim.x + threadIdx.x;
  int stride = gridDim.x * blockDim.x;
  if (tid == 0) { bar[0] = 0u; bar[1] = 0u; }
  for (int i = tid; i < BATCH * UNITS; i += stride) {
    h0f[i] = 0.0f; h1f[i] = 0.0f;
    h0A[i] = (_Float16)0.0f; h1A[i] = (_Float16)0.0f;
  }
}

// ---------------- gather emb column t and pack to f16 A layout (one 32B lane-chunk/thread) ---
// A 16x32 f16 layout (ISA 7.12.2): lane=(m%16)+16*half; slots 0..7 hold k=kb..kb+7,
// slots 8..15 hold k=kb+16..kb+23 with kb = kt*32 + 8*half.
__device__ __forceinline__ void pack_x(int tid, int t, const int* __restrict__ tokens,
                                       const float* __restrict__ emb, _Float16* __restrict__ xA) {
  for (int c = tid; c < MT * KT_X0 * 32; c += NTHREADS) {
    int lane2 = c & 31;
    int mtkt  = c >> 5;
    int mt = mtkt / KT_X0, kt = mtkt - mt * KT_X0;
    int m = mt * 16 + (lane2 & 15);
    int kb = kt * 32 + 8 * (lane2 >> 4);
    const float* er = emb + (size_t)tokens[m * SEQ + t] * HID;
    v16h outv;
    #pragma unroll
    for (int s = 0; s < 8; ++s) {
      int k0 = kb + s;
      int k1 = kb + 16 + s;
      outv[s]     = (_Float16)((k0 < HID) ? er[k0] : 0.0f);
      outv[s + 8] = (_Float16)((k1 < HID) ? er[k1] : 0.0f);
    }
    *(v16h*)(xA + ((size_t)c << 4)) = outv;
  }
}

// ---------------- gate GEMMs: G[64,2048] = [x@W | h@U] (+biases), one 16x16 tile per wave ----
__device__ __forceinline__ void gate_tiles(int wid, int lane,
    const _Float16* __restrict__ hA,           // recurrent A operand  [MT][KT_H][32][16]
    const _Float16* __restrict__ xAp, int ktx, // input A operand      [MT][ktx][32][16]
    const _Float16* __restrict__ Up,           // U packed             [NT][KT_H][32][16]
    const _Float16* __restrict__ Wp,           // W packed             [NT][ktx][32][16]
    const float* __restrict__ bias,            // [2][1536]
    float* __restrict__ G)                     // [64][2048]: z | r | xh | rh
{
  int j = wid;                                 // NWAVES == NT*MT: exactly one tile per wave
  if (j < NT * MT) {
    int nt = j >> 2;
    int mt = j & 3;
    int gcol = nt * 16 + (lane & 15);
    const v16h* Ah = (const v16h*)(hA  + (size_t)mt * KT_H * 512);
    const v16h* Bu = (const v16h*)(Up  + (size_t)nt * KT_H * 512);
    const v16h* Ax = (const v16h*)(xAp + (size_t)mt * ktx  * 512);
    const v16h* Bw = (const v16h*)(Wp  + (size_t)nt * ktx  * 512);
    if (nt < 64) {               // z and r gates: one fused accumulator
      v8f acc = {};
      #pragma unroll
      for (int kt = 0; kt < ktx; ++kt)
        acc = __builtin_amdgcn_wmma_f32_16x16x32_f16(false, Ax[kt * 32 + lane],
                                                     false, Bw[kt * 32 + lane],
                                                     (short)0, acc, false, false);
      #pragma unroll
      for (int kt = 0; kt < KT_H; ++kt)
        acc = __builtin_amdgcn_wmma_f32_16x16x32_f16(false, Ah[kt * 32 + lane],
                                                     false, Bu[kt * 32 + lane],
                                                     (short)0, acc, false, false);
      float bb = bias[gcol] + bias[G3 + gcol];
      #pragma unroll
      for (int i = 0; i < 8; ++i) {
        int m = mt * 16 + i + ((lane >> 4) << 3);
        G[(size_t)m * 2048 + gcol] = acc[i] + bb;
      }
    } else {                     // hh gate: keep x-part and recurrent part separate (reset_after)
      v8f accX = {};
      v8f accR = {};
      #pragma unroll
      for (int kt = 0; kt < ktx; ++kt)
        accX = __builtin_amdgcn_wmma_f32_16x16x32_f16(false, Ax[kt * 32 + lane],
                                                      false, Bw[kt * 32 + lane],
                                                      (short)0, accX, false, false);
      #pragma unroll
      for (int kt = 0; kt < KT_H; ++kt)
        accR = __builtin_amdgcn_wmma_f32_16x16x32_f16(false, Ah[kt * 32 + lane],
                                                      false, Bu[kt * 32 + lane],
                                                      (short)0, accR, false, false);
      float bi = bias[gcol], br = bias[G3 + gcol];
      #pragma unroll
      for (int i = 0; i < 8; ++i) {
        int m = mt * 16 + i + ((lane >> 4) << 3);
        G[(size_t)m * 2048 + gcol]       = accX[i] + bi;   // cols 1024..1535 : xh
        G[(size_t)m * 2048 + gcol + 512] = accR[i] + br;   // cols 1536..2047 : rh
      }
    }
  }
}

// ---------------- vectorized GRU update (+repack h to f16 A layout), 4 units/thread ----------
__device__ __forceinline__ void gru_update4(int tid, int t,
    const float* __restrict__ G, float* __restrict__ hf, _Float16* __restrict__ hA,
    float* __restrict__ yout, float* __restrict__ hlast)
{
  for (int e4 = tid; e4 < (BATCH * UNITS) / 4; e4 += NTHREADS) {
    int m   = e4 >> 7;               // 128 four-wide chunks per row
    int jj0 = (e4 & 127) << 2;
    const float* Gm = G + (size_t)m * 2048;
    float4 gz = *(const float4*)(Gm + jj0);
    float4 gr = *(const float4*)(Gm + 512 + jj0);
    float4 xh = *(const float4*)(Gm + 1024 + jj0);
    float4 rh = *(const float4*)(Gm + 1536 + jj0);
    float4 h  = *(const float4*)(hf + (size_t)m * UNITS + jj0);
    float4 hn;
    hn.x = gru1(gz.x, gr.x, xh.x, rh.x, h.x);
    hn.y = gru1(gz.y, gr.y, xh.y, rh.y, h.y);
    hn.z = gru1(gz.z, gr.z, xh.z, rh.z, h.z);
    hn.w = gru1(gz.w, gr.w, xh.w, rh.w, h.w);
    *(float4*)(hf + (size_t)m * UNITS + jj0) = hn;
    // pack into A layout: 4 contiguous jj stay in one group-of-8 -> single 8B store
    int g     = (jj0 & 31) >> 3;
    int lane2 = (m & 15) + ((g & 1) << 4);
    int s     = (jj0 & 7) + ((g >> 1) << 3);
    int mt = m >> 4, kt = jj0 >> 5;
    v4h hp = { (_Float16)hn.x, (_Float16)hn.y, (_Float16)hn.z, (_Float16)hn.w };
    *(v4h*)(hA + ((((size_t)mt * KT_H + kt) * 32 + lane2) << 4) + s) = hp;
    if (yout) {
      *(float4*)(yout + ((size_t)m * SEQ + t) * UNITS + jj0) = hn;
      if (t == SEQ - 1) *(float4*)(hlast + (size_t)m * UNITS + jj0) = hn;
    }
  }
}

// ---------------- persistent fused 2-layer GRU ----------------
__global__ void __launch_bounds__(TPB, 1) gru_persistent(
    const int* __restrict__ tokens, const float* __restrict__ emb,
    const float* __restrict__ b0, const float* __restrict__ b1,
    float* h0f, float* h1f, _Float16* h0A, _Float16* h1A, _Float16* xA,
    float* G0, float* G1,
    const _Float16* __restrict__ U0p, const _Float16* __restrict__ U1p,
    const _Float16* __restrict__ W0p, const _Float16* __restrict__ W1p,
    unsigned* bar, float* __restrict__ out)
{
  const int tid  = blockIdx.x * blockDim.x + threadIdx.x;
  const int wid  = tid >> 5;
  const int lane = threadIdx.x & 31;
  float* y1    = out;
  float* h1out = out + (size_t)BATCH * SEQ * UNITS;

  // prologue: pack x_0
  pack_x(tid, 0, tokens, emb, xA);
  gbar(bar);

  for (int t = 0; t < SEQ; ++t) {
    // P1: layer-0 gates  G0 = x_t@W0 + h0@U0 (+b0)
    gate_tiles(wid, lane, h0A, xA, KT_X0, U0p, W0p, b0, G0);
    gbar(bar);
    // P2: layer-0 update (h0 master copy in f32, repacked f16)
    gru_update4(tid, t, G0, h0f, h0A, nullptr, nullptr);
    gbar(bar);
    // P3: layer-1 gates  G1 = y0_t@W1 + h1@U1 (+b1); y0_t == h0 (packed)
    gate_tiles(wid, lane, h1A, h0A, KT_H, U1p, W1p, b1, G1);
    gbar(bar);
    // P4: layer-1 update + emit y1[:,t,:] (and h_last at t==S-1); also prefetch-pack x_{t+1}
    gru_update4(tid, t, G1, h1f, h1A, y1, h1out);
    if (t + 1 < SEQ) pack_x(tid, t + 1, tokens, emb, xA);
    gbar(bar);
  }
}

// ---------------- host ----------------
extern "C" void kernel_launch(void* const* d_in, const int* in_sizes, int n_in,
                              void* d_out, int out_size, void* d_ws, size_t ws_size,
                              hipStream_t stream) {
  const int*   tokens = (const int*)  d_in[0];
  const float* emb    = (const float*)d_in[1];
  const float* W0     = (const float*)d_in[2];
  const float* U0     = (const float*)d_in[3];
  const float* b0     = (const float*)d_in[4];
  const float* W1     = (const float*)d_in[5];
  const float* U1     = (const float*)d_in[6];
  const float* b1     = (const float*)d_in[7];
  float* out = (float*)d_out;

  char* w = (char*)d_ws;
  size_t off = 0;
  auto take = [&](size_t bytes) -> char* {
    char* p = w + off;
    off += (bytes + 255) & ~(size_t)255;
    return p;
  };
  float*     h0f = (float*)    take((size_t)BATCH * UNITS * 4);
  float*     h1f = (float*)    take((size_t)BATCH * UNITS * 4);
  _Float16*  h0A = (_Float16*) take((size_t)MT * KT_H  * 512 * 2);
  _Float16*  h1A = (_Float16*) take((size_t)MT * KT_H  * 512 * 2);
  _Float16*  xA  = (_Float16*) take((size_t)MT * KT_X0 * 512 * 2);
  float*     G0  = (float*)    take((size_t)BATCH * 2048 * 4);
  float*     G1  = (float*)    take((size_t)BATCH * 2048 * 4);
  _Float16*  U0p = (_Float16*) take((size_t)NT * KT_H  * 512 * 2);
  _Float16*  U1p = (_Float16*) take((size_t)NT * KT_H  * 512 * 2);
  _Float16*  W1p = (_Float16*) take((size_t)NT * KT_H  * 512 * 2);
  _Float16*  W0p = (_Float16*) take((size_t)NT * KT_X0 * 512 * 2);
  unsigned*  bar = (unsigned*) take(2 * sizeof(unsigned));

  pack_weights<<<768, 256, 0, stream>>>(W0, W0p, HID,   KT_X0);
  pack_weights<<<768, 256, 0, stream>>>(U0, U0p, UNITS, KT_H);
  pack_weights<<<768, 256, 0, stream>>>(W1, W1p, UNITS, KT_H);
  pack_weights<<<768, 256, 0, stream>>>(U1, U1p, UNITS, KT_H);
  init_state<<<64, 256, 0, stream>>>(h0f, h1f, h0A, h1A, bar);
  gru_persistent<<<NWG, TPB, 0, stream>>>(tokens, emb, b0, b1,
                                          h0f, h1f, h0A, h1A, xA, G0, G1,
                                          U0p, U1p, W0p, W1p, bar, out);
}
